// SelfAttention_87325275062363
// MI455X (gfx1250) — matrix operand order repaired and checked by reading, not compiled, not run on hardware
//
#include <hip/hip_runtime.h>
#include <hip/hip_bf16.h>

typedef __attribute__((ext_vector_type(16))) __bf16 bf16x16;
typedef __attribute__((ext_vector_type(8)))  float  f32x8;
typedef __attribute__((ext_vector_type(4)))  int    v4i_t;

typedef __attribute__((address_space(3))) char lds_char_t;
typedef __attribute__((address_space(1))) char gbl_char_t;
typedef __attribute__((address_space(1))) v4i_t g_v4i;
typedef __attribute__((address_space(3))) v4i_t l_v4i;

#define WMMA_BF16(a, b, c) \
  __builtin_amdgcn_wmma_f32_16x16x32_bf16(false, (a), false, (b), (short)0, (c), false, false)

// ---- CDNA5 async global->LDS copy (ASYNCcnt path), 16 bytes per lane -------
__device__ __forceinline__ void async_ld16(void* lds, const void* g) {
#if __has_builtin(__builtin_amdgcn_global_load_async_to_lds_b128)
  g_v4i* gp = (g_v4i*)(gbl_char_t*)(const char*)g;
  l_v4i* lp = (l_v4i*)(lds_char_t*)(char*)lds;
  __builtin_amdgcn_global_load_async_to_lds_b128(gp, lp, 0, 0);
#else
  unsigned loff = (unsigned)(__SIZE_TYPE__)(lds_char_t*)(char*)lds;
  unsigned long long ga = (unsigned long long)(const char*)g;
  asm volatile("global_load_async_to_lds_b128 %0, %1, off"
               :: "v"(loff), "v"(ga) : "memory");
#endif
}

__device__ __forceinline__ void wait_async0() {
#if __has_builtin(__builtin_amdgcn_s_wait_asynccnt)
  __builtin_amdgcn_s_wait_asynccnt(0);
#else
  asm volatile("s_wait_asynccnt 0x0" ::: "memory");
#endif
}

// ---------------------------------------------------------------- cast kernel
__global__ void f32_to_bf16(const float4* __restrict__ in, __bf16* __restrict__ out, int n4) {
  int i = blockIdx.x * blockDim.x + threadIdx.x;
  int stride = gridDim.x * blockDim.x;
  for (; i < n4; i += stride) {
    float4 f = in[i];
    alignas(8) __bf16 p[4] = {(__bf16)f.x, (__bf16)f.y, (__bf16)f.z, (__bf16)f.w};
    *(uint2*)(out + 4 * (size_t)i) = *(const uint2*)p;
  }
}

// ---------------------------------------------------------------- QKV GEMM
// X[8192,1024] bf16 @ Wqkv^T -> Q/K (b,h,s,dh) and V transposed (b,h,dh,s), all bf16
__global__ __launch_bounds__(256) void qkv_gemm(
    const __bf16* __restrict__ xbf,   // [8192,1024]
    const __bf16* __restrict__ wbf,   // [3072,1024]
    const float*  __restrict__ bqkv,  // [3072]
    __bf16* __restrict__ qbf, __bf16* __restrict__ kbf, __bf16* __restrict__ vtbf) {
  __shared__ __bf16 As[2][128][40];   // double-buffered 128x32 tile (pad 8)
  __shared__ __bf16 Bst[2][64][40];   // double-buffered 64x32 transposed W tile
  const int t = threadIdx.x;
  const int lane = t & 31, w = t >> 5;
  const int g = lane >> 4, mrow = lane & 15;
  const int mbase = blockIdx.y * 128;
  const int nbase = blockIdx.x * 64;

  const int li = t >> 1, lc = (t & 1) * 16;     // As loader mapping
  const int ln = t >> 2, ls = (t & 3) * 8;      // Bst loader mapping

  auto issue = [&](int k0, int buf) {
    const __bf16* s = xbf + (size_t)(mbase + li) * 1024 + k0 + lc;
    async_ld16(&As[buf][li][lc], s);
    async_ld16(&As[buf][li][lc + 8], s + 8);
    async_ld16(&Bst[buf][ln][ls], wbf + (size_t)(nbase + ln) * 1024 + k0 + ls);
  };

  f32x8 acc[4] = {};
  issue(0, 0);

  for (int it = 0; it < 32; ++it) {
    const int cur = it & 1;
    wait_async0();
    __syncthreads();
    if (it + 1 < 32) issue((it + 1) * 32, cur ^ 1);

    bf16x16 a;
#pragma unroll
    for (int v = 0; v < 8; ++v) {
      int kk = ((v < 4) ? 2 * v : 2 * v + 8) + 8 * g;  // A 16-bit 16x32 layout
      a[2 * v]     = As[cur][w * 16 + mrow][kk];
      a[2 * v + 1] = As[cur][w * 16 + mrow][kk + 1];
    }
#pragma unroll
    for (int nt = 0; nt < 4; ++nt) {
      bf16x16 bf;
#pragma unroll
      for (int v = 0; v < 8; ++v) {
        int kk = ((v < 4) ? 2 * v : 2 * v + 8) + 8 * g;  // B 32x16 layout (same k order as A)
        bf[2 * v]     = Bst[cur][nt * 16 + mrow][kk];
        bf[2 * v + 1] = Bst[cur][nt * 16 + mrow][kk + 1];
      }
      acc[nt] = WMMA_BF16(a, bf, acc[nt]);
    }
  }

  // epilogue: block-uniform section/head/batch, hoisted
  const int sec = nbase >> 10;                 // 0=Q 1=K 2=V
  const int h   = (nbase & 1023) >> 6;
  const int b   = mbase >> 11;
  const int sbase = (mbase & 2047) + w * 16 + 8 * g;   // + v per element

  if (sec == 2) {
    __bf16* vb = vtbf + (size_t)(b * 16 + h) * 64 * 2048;
#pragma unroll
    for (int nt = 0; nt < 4; ++nt) {
      int dh = nt * 16 + mrow;
      float bias = bqkv[nbase + nt * 16 + mrow];
      alignas(16) __bf16 pk[8];
#pragma unroll
      for (int v = 0; v < 8; ++v) pk[v] = (__bf16)(acc[nt][v] + bias);
      *(uint4*)(vb + (size_t)dh * 2048 + sbase) = *(const uint4*)pk;  // 8 contiguous s
    }
  } else {
    __bf16* dst = (sec == 0 ? qbf : kbf) + (size_t)(b * 16 + h) * 2048 * 64;
#pragma unroll
    for (int nt = 0; nt < 4; ++nt) {
      int dh = nt * 16 + mrow;
      float bias = bqkv[nbase + nt * 16 + mrow];
#pragma unroll
      for (int v = 0; v < 8; ++v)
        dst[(size_t)(sbase + v) * 64 + dh] = (__bf16)(acc[nt][v] + bias);
    }
  }
}

// ---------------------------------------------------------------- flash attention
// one wave = 16 query rows; scores computed transposed: S^T = K . Q^T, O^T = V^T . P
__global__ __launch_bounds__(256) void attn_kernel(
    const __bf16* __restrict__ qbf, const __bf16* __restrict__ kbf,
    const __bf16* __restrict__ vtbf, const int* __restrict__ key_mask,
    __bf16* __restrict__ attnbf) {
  __shared__ __bf16 Ks[2][32][72];    // 32 keys x 64 dh, double-buffered
  __shared__ __bf16 Vts[2][64][40];   // 64 dh x 32 keys, double-buffered
  __shared__ float maskb[2][32];
  const int t = threadIdx.x, lane = t & 31, w = t >> 5;
  const int g = lane >> 4, nq = lane & 15;
  const int bh = blockIdx.y, b = bh >> 4, h = bh & 15;
  const int qbase = blockIdx.x * 128 + w * 16;
  const float slope = exp2f(-8.0f * (float)(h + 1) / 16.0f);
  const float qposf = (float)(qbase + nq);

  const __bf16* kp = kbf + (size_t)(b * 16 + h) * 2048 * 64;
  const __bf16* vp = vtbf + (size_t)(b * 16 + h) * 64 * 2048;

  const int kr = t >> 3, ksc = (t & 7) * 8;    // Ks loader mapping
  const int dv = t >> 2, vsc = (t & 3) * 8;    // Vts loader mapping

  auto issue = [&](int kc, int buf) {
    async_ld16(&Ks[buf][kr][ksc], kp + (size_t)(kc + kr) * 64 + ksc);
    async_ld16(&Vts[buf][dv][vsc], vp + (size_t)dv * 2048 + kc + vsc);
    if (t < 32) maskb[buf][t] = key_mask[b * 2048 + kc + t] ? 0.0f : -1e30f;
  };

  // Q^T fragments (B-matrix, dh x q), built once per wave
  const __bf16* qrow = qbf + ((size_t)(b * 16 + h) * 2048 + (qbase + nq)) * 64;
  bf16x16 bq[2];
#pragma unroll
  for (int c = 0; c < 2; ++c)
#pragma unroll
    for (int v = 0; v < 8; ++v) {
      int dh = 32 * c + ((v < 4) ? 2 * v : 2 * v + 8) + 8 * g;
      bq[c][2 * v]     = qrow[dh];
      bq[c][2 * v + 1] = qrow[dh + 1];
    }

  f32x8 o[4] = {};
  float mrow = -1e30f, lrow = 0.0f;

  issue(0, 0);

  for (int it = 0; it < 64; ++it) {
    const int cur = it & 1;
    const int kc = it * 32;
    wait_async0();
    __syncthreads();
    if (it + 1 < 64) issue(kc + 32, cur ^ 1);

    // S^T tiles: St0 = keys kc..kc+15, St1 = keys kc+16..kc+31 (C-layout lane = query)
    f32x8 st0 = {}, st1 = {};
#pragma unroll
    for (int c = 0; c < 2; ++c) {
      bf16x16 ak;
#pragma unroll
      for (int v = 0; v < 8; ++v) {
        int dh = ((v < 4) ? 2 * v : 2 * v + 8) + 8 * g + 32 * c;
        ak[2 * v]     = Ks[cur][nq][dh];
        ak[2 * v + 1] = Ks[cur][nq][dh + 1];
      }
      st0 = WMMA_BF16(ak, bq[c], st0);
#pragma unroll
      for (int v = 0; v < 8; ++v) {
        int dh = ((v < 4) ? 2 * v : 2 * v + 8) + 8 * g + 32 * c;
        ak[2 * v]     = Ks[cur][16 + nq][dh];
        ak[2 * v + 1] = Ks[cur][16 + nq][dh + 1];
      }
      st1 = WMMA_BF16(ak, bq[c], st1);
    }

    // scale + ALiBi + mask; online softmax (row = query = this lane)
    float s0[8], s1[8], cmax = -1e30f;
#pragma unroll
    for (int v = 0; v < 8; ++v) {
      int k0l = 8 * g + v, k1l = 16 + 8 * g + v;
      s0[v] = st0[v] * 0.125f + maskb[cur][k0l] - fabsf(qposf - (float)(kc + k0l)) * slope;
      s1[v] = st1[v] * 0.125f + maskb[cur][k1l] - fabsf(qposf - (float)(kc + k1l)) * slope;
      cmax = fmaxf(cmax, fmaxf(s0[v], s1[v]));
    }
    cmax = fmaxf(cmax, __shfl_xor(cmax, 16, 32));
    float mnew = fmaxf(mrow, cmax);
    float alpha = __expf(mrow - mnew);
    mrow = mnew;

    float p0[8], p1[8], lsum = 0.0f;
#pragma unroll
    for (int v = 0; v < 8; ++v) {
      p0[v] = __expf(s0[v] - mnew);
      p1[v] = __expf(s1[v] - mnew);
      lsum += p0[v] + p1[v];
    }
    lsum += __shfl_xor(lsum, 16, 32);
    lrow = lrow * alpha + lsum;
#pragma unroll
    for (int tt = 0; tt < 4; ++tt)
#pragma unroll
      for (int v = 0; v < 8; ++v) o[tt][v] *= alpha;

    // P fragment (B-matrix 32x16: lane = query, element e <-> k = 16g+e)
    bf16x16 bp;
#pragma unroll
    for (int v = 0; v < 8; ++v) {
      float q0 = __shfl_xor(p0[v], 16, 32);
      float q1 = __shfl_xor(p1[v], 16, 32);
      bp[v]     = (__bf16)(g ? q1 : p0[v]);
      bp[8 + v] = (__bf16)(g ? p1[v] : q0);
    }

    // O^T += V^T . P  (4 dh tiles of 16)
#pragma unroll
    for (int tt = 0; tt < 4; ++tt) {
      bf16x16 av;
#pragma unroll
      for (int v = 0; v < 8; ++v) {
        int kk = 2 * v + 16 * g;
        av[2 * v]     = Vts[cur][tt * 16 + nq][kk];
        av[2 * v + 1] = Vts[cur][tt * 16 + nq][kk + 1];
      }
      o[tt] = WMMA_BF16(av, bp, o[tt]);
    }
  }

  // normalize and store attn output bf16 [b, s, h*64+dh]
  float inv = 1.0f / lrow;
  __bf16* orow = attnbf + ((size_t)b * 2048 + (qbase + nq)) * 1024 + h * 64;
#pragma unroll
  for (int tt = 0; tt < 4; ++tt) {
    alignas(16) __bf16 pk[8];
#pragma unroll
    for (int v = 0; v < 8; ++v) pk[v] = (__bf16)(o[tt][v] * inv);  // dh = 16tt + 8g + v
    *(uint4*)(orow + 16 * tt + 8 * g) = *(const uint4*)pk;
  }
}

// ---------------------------------------------------------------- output GEMM
__global__ __launch_bounds__(256) void out_gemm(
    const __bf16* __restrict__ abf,   // [8192,1024]
    const __bf16* __restrict__ wbf,   // [1024,1024]
    const float*  __restrict__ bout,  // [1024]
    float* __restrict__ out) {
  __shared__ __bf16 As[2][128][40];
  __shared__ __bf16 Bst[2][64][40];
  const int t = threadIdx.x;
  const int lane = t & 31, w = t >> 5;
  const int g = lane >> 4, mrow = lane & 15;
  const int mbase = blockIdx.y * 128;
  const int nbase = blockIdx.x * 64;

  const int li = t >> 1, lc = (t & 1) * 16;
  const int ln = t >> 2, ls = (t & 3) * 8;

  auto issue = [&](int k0, int buf) {
    const __bf16* s = abf + (size_t)(mbase + li) * 1024 + k0 + lc;
    async_ld16(&As[buf][li][lc], s);
    async_ld16(&As[buf][li][lc + 8], s + 8);
    async_ld16(&Bst[buf][ln][ls], wbf + (size_t)(nbase + ln) * 1024 + k0 + ls);
  };

  f32x8 acc[4] = {};
  issue(0, 0);

  for (int it = 0; it < 32; ++it) {
    const int cur = it & 1;
    wait_async0();
    __syncthreads();
    if (it + 1 < 32) issue((it + 1) * 32, cur ^ 1);

    bf16x16 a;
#pragma unroll
    for (int v = 0; v < 8; ++v) {
      int kk = ((v < 4) ? 2 * v : 2 * v + 8) + 8 * g;
      a[2 * v]     = As[cur][w * 16 + mrow][kk];
      a[2 * v + 1] = As[cur][w * 16 + mrow][kk + 1];
    }
#pragma unroll
    for (int nt = 0; nt < 4; ++nt) {
      bf16x16 bf;
#pragma unroll
      for (int v = 0; v < 8; ++v) {
        int kk = ((v < 4) ? 2 * v : 2 * v + 8) + 8 * g;
        bf[2 * v]     = Bst[cur][nt * 16 + mrow][kk];
        bf[2 * v + 1] = Bst[cur][nt * 16 + mrow][kk + 1];
      }
      acc[nt] = WMMA_BF16(a, bf, acc[nt]);
    }
  }

#pragma unroll
  for (int nt = 0; nt < 4; ++nt) {
    int c = nbase + nt * 16 + mrow;
    float bias = bout[c];
#pragma unroll
    for (int v = 0; v < 8; ++v) {
      int r = mbase + w * 16 + v + 8 * g;
      out[(size_t)r * 1024 + c] = acc[nt][v] + bias;
    }
  }
}

// ---------------------------------------------------------------- launch
extern "C" void kernel_launch(void* const* d_in, const int* in_sizes, int n_in,
                              void* d_out, int out_size, void* d_ws, size_t ws_size,
                              hipStream_t stream) {
  const float* x        = (const float*)d_in[0];
  const int*   key_mask = (const int*)d_in[1];
  const float* Wqkv     = (const float*)d_in[2];
  const float* bqkv     = (const float*)d_in[3];
  const float* Wout     = (const float*)d_in[4];
  const float* bout     = (const float*)d_in[5];
  float* out = (float*)d_out;

  char* ws = (char*)d_ws;
  __bf16* xbf    = (__bf16*)ws; ws += (size_t)8192 * 1024 * 2;
  __bf16* wqkvbf = (__bf16*)ws; ws += (size_t)3072 * 1024 * 2;
  __bf16* woutbf = (__bf16*)ws; ws += (size_t)1024 * 1024 * 2;
  __bf16* qbf    = (__bf16*)ws; ws += (size_t)8192 * 1024 * 2;
  __bf16* kbf    = (__bf16*)ws; ws += (size_t)8192 * 1024 * 2;
  __bf16* vtbf   = (__bf16*)ws; ws += (size_t)8192 * 1024 * 2;
  __bf16* attnbf = (__bf16*)ws; ws += (size_t)8192 * 1024 * 2;

  f32_to_bf16<<<1024, 256, 0, stream>>>((const float4*)x, xbf, 8192 * 1024 / 4);
  f32_to_bf16<<<512, 256, 0, stream>>>((const float4*)Wqkv, wqkvbf, 3072 * 1024 / 4);
  f32_to_bf16<<<256, 256, 0, stream>>>((const float4*)Wout, woutbf, 1024 * 1024 / 4);

  qkv_gemm<<<dim3(48, 64), 256, 0, stream>>>(xbf, wqkvbf, bqkv, qbf, kbf, vtbf);
  attn_kernel<<<dim3(16, 64), 256, 0, stream>>>(qbf, kbf, vtbf, key_mask, attnbf);
  out_gemm<<<dim3(16, 64), 256, 0, stream>>>(attnbf, woutbf, bout, out);
}
